// TransformerBlock_18545668784207
// MI455X (gfx1250) — matrix operand order repaired and checked by reading, not compiled, not run on hardware
//
#include <hip/hip_runtime.h>
#include <hip/hip_bf16.h>

// ---- dims ----
#define BATCH   4
#define SEQ     2048
#define DMODEL  1024
#define NHEADS  16
#define HDEPTH  64
#define MTOK    (BATCH*SEQ)       // 8192
#define KDIM    1024
#define NDIM    1024

typedef __attribute__((ext_vector_type(16))) __bf16 v16bf;
typedef __attribute__((ext_vector_type(8)))  __bf16 v8bf;
typedef __attribute__((ext_vector_type(8)))  float  v8f;

union BF16x16 { v16bf v; v8bf h[2]; };

typedef __attribute__((address_space(3))) __bf16 lds_bf16_t;

__device__ __forceinline__ v8f wmma_bf16(v16bf a, v16bf b, v8f c) {
  // D = A(16x32 bf16) * B(32x16 bf16) + C(16x16 f32)
  return __builtin_amdgcn_wmma_f32_16x16x32_bf16(
      /*neg_a=*/false, a, /*neg_b=*/false, b,
      /*c_mod=*/(short)0, c, /*reuse_a=*/false, /*reuse_b=*/false);
}

// CDNA5 async global->LDS copy (16B per lane), tracked with ASYNCcnt.
__device__ __forceinline__ void async_g2l_b128(const __bf16* gptr, __bf16* lptr) {
  unsigned lds_off = (unsigned)(size_t)(lds_bf16_t*)lptr;
  unsigned long long gva = (unsigned long long)(size_t)gptr;
  asm volatile("global_load_async_to_lds_b128 %0, %1, off"
               :: "v"(lds_off), "v"(gva) : "memory");
}
__device__ __forceinline__ void wait_asynccnt0() {
  asm volatile("s_wait_asynccnt 0x0" ::: "memory");
}

// ---------------------------------------------------------------------------
// Weight fp32 [K][N] -> bf16 transposed [N][K]
// ---------------------------------------------------------------------------
__global__ __launch_bounds__(256)
void transpose_w_bf16(const float* __restrict__ W, __bf16* __restrict__ Wt) {
  int idx = blockIdx.x * 256 + threadIdx.x;       // over K*N
  int k = idx >> 10, n = idx & 1023;              // coalesced read over n
  Wt[n * KDIM + k] = (__bf16)W[k * NDIM + n];
}

// ---------------------------------------------------------------------------
// x = inputs + positional encoding ; write fp32 residual + bf16 operand
// ---------------------------------------------------------------------------
__global__ __launch_bounds__(256)
void add_pe(const float* __restrict__ in, float* __restrict__ xf,
            __bf16* __restrict__ xb) {
  int idx = blockIdx.x * 256 + threadIdx.x;       // over B*S*D
  int d = idx & 1023;
  int s = (idx >> 10) & (SEQ - 1);
  float i2   = (float)((d >> 1) << 1);
  float rate = __powf(10000.0f, -i2 * (1.0f / (float)DMODEL));
  float ang  = (float)s * rate;
  float pe   = (d & 1) ? __cosf(ang) : __sinf(ang);
  float v = in[idx] + pe;
  xf[idx] = v;
  xb[idx] = (__bf16)v;
}

// ---------------------------------------------------------------------------
// Generic bf16 WMMA GEMM:  C[M,N] = A[M,K] @ Wt[N,K]^T + bias
// BM=128 BN=64 BK=32, 256 threads (8 waves), each wave = 32x32 (2x2 WMMA tiles)
// Tiles staged with CDNA5 GLOBAL_LOAD_ASYNC_TO_LDS_B128 (no VGPR round-trip).
// ---------------------------------------------------------------------------
constexpr int EPI_F32       = 0;  // fp32 out [M,N]
constexpr int EPI_HEADS     = 1;  // bf16 out scattered [B,H,S,64]   (Q,K)
constexpr int EPI_HEADS_T   = 2;  // bf16 out scattered [B,H,64,S]   (V)
constexpr int EPI_RELU_BF16 = 3;  // relu -> bf16 out [M,N]          (FFN1)

template<int MODE>
__global__ __launch_bounds__(256)
void gemm_bf16_wmma(const __bf16* __restrict__ A, const __bf16* __restrict__ Wt,
                    const float* __restrict__ bias, void* __restrict__ out) {
  __shared__ __bf16 As[128 * 32];
  __shared__ __bf16 Bs[64 * 32];

  const int tid  = threadIdx.x;
  const int lane = tid & 31, wave = tid >> 5;
  const int half = lane >> 4, l16 = lane & 15;
  const int wm = (wave & 3) * 32;          // wave M offset in tile
  const int wn = (wave >> 2) * 32;         // wave N offset in tile
  const int rowBase = blockIdx.y * 128;
  const int nBase   = blockIdx.x * 64;

  // staging coordinates (fixed per thread)
  const int ar0 = (tid * 2) >> 2,     ac0 = ((tid * 2) & 3) * 8;
  const int ar1 = (tid * 2 + 1) >> 2, ac1 = ((tid * 2 + 1) & 3) * 8;
  const int br  = tid >> 2,           bc  = (tid & 3) * 8;

  v8f acc[2][2] = {};

  for (int k0 = 0; k0 < KDIM; k0 += 32) {
    // async stage A tile (128x32 bf16) and B tile (64x32 bf16) straight to LDS
    async_g2l_b128(&A[(size_t)(rowBase + ar0) * KDIM + k0 + ac0], &As[ar0 * 32 + ac0]);
    async_g2l_b128(&A[(size_t)(rowBase + ar1) * KDIM + k0 + ac1], &As[ar1 * 32 + ac1]);
    async_g2l_b128(&Wt[(size_t)(nBase + br) * KDIM + k0 + bc],    &Bs[br * 32 + bc]);
    if (k0 + 32 < KDIM)  // hint next A tile toward L2 (global_prefetch_b8)
      __builtin_prefetch(&A[(size_t)(rowBase + ar0) * KDIM + k0 + 32], 0, 1);
    wait_asynccnt0();
    __syncthreads();

    // A fragments (16x32 bf16: row = l16, half selects K half, 16 contiguous)
    BF16x16 af[2], bf[2];
#pragma unroll
    for (int mi = 0; mi < 2; ++mi) {
      int r = wm + mi * 16 + l16;
      af[mi].v = *(const v16bf*)&As[r * 32 + half * 16];
    }
    // B fragments (32x16: column = l16, half selects K half, 16 contiguous)
#pragma unroll
    for (int ni = 0; ni < 2; ++ni) {
      int c = wn + ni * 16 + l16;
      bf[ni].v = *(const v16bf*)&Bs[c * 32 + half * 16];
    }
#pragma unroll
    for (int mi = 0; mi < 2; ++mi)
#pragma unroll
      for (int ni = 0; ni < 2; ++ni)
        acc[mi][ni] = wmma_bf16(af[mi].v, bf[ni].v, acc[mi][ni]);
    __syncthreads();
  }

  // epilogue: C layout -> lane holds col=l16, rows half*8+r
#pragma unroll
  for (int mi = 0; mi < 2; ++mi) {
#pragma unroll
    for (int ni = 0; ni < 2; ++ni) {
      int gn = nBase + wn + ni * 16 + l16;
      float bb = bias[gn];
#pragma unroll
      for (int r = 0; r < 8; ++r) {
        int gm = rowBase + wm + mi * 16 + half * 8 + r;
        float v = acc[mi][ni][r] + bb;
        if (MODE == EPI_F32) {
          ((float*)out)[(size_t)gm * NDIM + gn] = v;
        } else if (MODE == EPI_RELU_BF16) {
          ((__bf16*)out)[(size_t)gm * NDIM + gn] = (__bf16)fmaxf(v, 0.0f);
        } else if (MODE == EPI_HEADS) {
          int b_ = gm >> 11, s_ = gm & (SEQ - 1);
          int h_ = gn >> 6,  d_ = gn & 63;
          ((__bf16*)out)[((size_t)((b_ * NHEADS + h_) * SEQ) + s_) * HDEPTH + d_] =
              (__bf16)v;
        } else { // EPI_HEADS_T
          int b_ = gm >> 11, s_ = gm & (SEQ - 1);
          int h_ = gn >> 6,  d_ = gn & 63;
          ((__bf16*)out)[((size_t)((b_ * NHEADS + h_) * HDEPTH) + d_) * SEQ + s_] =
              (__bf16)v;
        }
      }
    }
  }
}

// ---------------------------------------------------------------------------
// Flash-style attention. Block = 8 waves, each wave owns 16 query rows,
// streams keys in chunks of 32. Q[B,H,S,64], K[B,H,S,64], VT[B,H,64,S].
// grid = (B*H) * (S/128)
// ---------------------------------------------------------------------------
__global__ __launch_bounds__(256)
void attention_kernel(const __bf16* __restrict__ Q, const __bf16* __restrict__ K,
                      const __bf16* __restrict__ VT, __bf16* __restrict__ ctx) {
  __shared__ __bf16 Ps[8 * 16 * 32];   // per-wave 16x32 P tile for C->A relayout
  const int tid = threadIdx.x, lane = tid & 31, wave = tid >> 5;
  const int half = lane >> 4, l16 = lane & 15;
  const int bh = blockIdx.x >> 4;          // b*H + h
  const int qb = blockIdx.x & 15;
  const int b_ = bh >> 4, h_ = bh & 15;
  const int qrow0 = qb * 128 + wave * 16;

  const __bf16* qh  = Q  + (size_t)bh * SEQ * HDEPTH;
  const __bf16* kh  = K  + (size_t)bh * SEQ * HDEPTH;
  const __bf16* vth = VT + (size_t)bh * HDEPTH * SEQ;
  __bf16* Pw = &Ps[wave * 16 * 32];

  // Q A-fragments for both depth k-steps (depth 64 = 2 x K32)
  BF16x16 aq[2];
  {
    int row = qrow0 + l16;
#pragma unroll
    for (int ks = 0; ks < 2; ++ks) {
      aq[ks].v = *(const v16bf*)&qh[(size_t)row * HDEPTH + ks * 32 + half * 16];
    }
  }

  float mrow[8], lrow[8];
  v8f accc[4] = {};
#pragma unroll
  for (int r = 0; r < 8; ++r) { mrow[r] = -1e30f; lrow[r] = 0.0f; }

  for (int kc = 0; kc < SEQ; kc += 32) {
    // S = Q @ K^T for 32 keys (2 n-tiles x 2 depth k-steps = 4 WMMAs)
    v8f s[2] = {};
#pragma unroll
    for (int ni = 0; ni < 2; ++ni) {
      int key = kc + ni * 16 + l16;
#pragma unroll
      for (int ks = 0; ks < 2; ++ks) {
        v16bf bk = *(const v16bf*)&kh[(size_t)key * HDEPTH + ks * 32 + half * 16];
        s[ni] = wmma_bf16(aq[ks].v, bk, s[ni]);
      }
    }
    // online softmax (row stats via 16-lane xor reductions; 1/sqrt(64)=0.125)
#pragma unroll
    for (int r = 0; r < 8; ++r) {
      float s0 = s[0][r] * 0.125f, s1 = s[1][r] * 0.125f;
      float t = fmaxf(s0, s1);
#pragma unroll
      for (int msk = 8; msk >= 1; msk >>= 1) t = fmaxf(t, __shfl_xor(t, msk, 32));
      float mn = fmaxf(mrow[r], t);
      float p0 = __expf(s0 - mn), p1 = __expf(s1 - mn);
      float ps = p0 + p1;
#pragma unroll
      for (int msk = 8; msk >= 1; msk >>= 1) ps += __shfl_xor(ps, msk, 32);
      float alpha = __expf(mrow[r] - mn);
      lrow[r] = lrow[r] * alpha + ps;
      mrow[r] = mn;
#pragma unroll
      for (int ni = 0; ni < 4; ++ni) accc[ni][r] *= alpha;
      s[0][r] = p0; s[1][r] = p1;
    }
    // P: C-layout -> LDS -> A-fragment layout (per-wave slice, no barrier)
#pragma unroll
    for (int t = 0; t < 2; ++t)
#pragma unroll
      for (int r = 0; r < 8; ++r)
        Pw[(half * 8 + r) * 32 + t * 16 + l16] = (__bf16)s[t][r];
    BF16x16 ap;
    ap.v = *(const v16bf*)&Pw[l16 * 32 + half * 16];
    // ctx += P @ V  (depth 64 = 4 n-tiles)
#pragma unroll
    for (int ni = 0; ni < 4; ++ni) {
      int d = ni * 16 + l16;
      v16bf bv = *(const v16bf*)&vth[(size_t)d * SEQ + kc + half * 16];
      accc[ni] = wmma_bf16(ap.v, bv, accc[ni]);
    }
  }

  // normalize rows, scatter back to [B,S,D] bf16
#pragma unroll
  for (int r = 0; r < 8; ++r) {
    float inv = 1.0f / lrow[r];
    int srow = qrow0 + half * 8 + r;
    size_t base = ((size_t)(b_ * SEQ + srow)) * DMODEL + h_ * HDEPTH;
#pragma unroll
    for (int ni = 0; ni < 4; ++ni)
      ctx[base + ni * 16 + l16] = (__bf16)(accc[ni][r] * inv);
  }
}

// ---------------------------------------------------------------------------
// out = LayerNorm(X + Yadd) * g + b  ; fp32 out + optional bf16 copy
// one 256-thread block per 1024-wide row
// ---------------------------------------------------------------------------
__global__ __launch_bounds__(256)
void residual_ln(const float* __restrict__ X, const float* __restrict__ Yadd,
                 const float* __restrict__ g, const float* __restrict__ bta,
                 float* __restrict__ outf, __bf16* __restrict__ outb) {
  __shared__ float red[256];
  const int row = blockIdx.x, tid = threadIdx.x;
  const float* xr = X    + (size_t)row * DMODEL;
  const float* yr = Yadd + (size_t)row * DMODEL;
  float v[4]; float sum = 0.0f;
#pragma unroll
  for (int i = 0; i < 4; ++i) { v[i] = xr[tid + i * 256] + yr[tid + i * 256]; sum += v[i]; }
  red[tid] = sum; __syncthreads();
  for (int s2 = 128; s2 > 0; s2 >>= 1) { if (tid < s2) red[tid] += red[tid + s2]; __syncthreads(); }
  float mu = red[0] * (1.0f / (float)DMODEL);
  __syncthreads();
  float sq = 0.0f;
#pragma unroll
  for (int i = 0; i < 4; ++i) { float d = v[i] - mu; sq += d * d; }
  red[tid] = sq; __syncthreads();
  for (int s2 = 128; s2 > 0; s2 >>= 1) { if (tid < s2) red[tid] += red[tid + s2]; __syncthreads(); }
  float rstd = rsqrtf(red[0] * (1.0f / (float)DMODEL) + 1e-6f);
#pragma unroll
  for (int i = 0; i < 4; ++i) {
    int c = tid + i * 256;
    float o = (v[i] - mu) * rstd * g[c] + bta[c];
    outf[(size_t)row * DMODEL + c] = o;
    if (outb) outb[(size_t)row * DMODEL + c] = (__bf16)o;
  }
}

// ---------------------------------------------------------------------------
extern "C" void kernel_launch(void* const* d_in, const int* in_sizes, int n_in,
                              void* d_out, int out_size, void* d_ws, size_t ws_size,
                              hipStream_t stream) {
  const float* inputs = (const float*)d_in[0];
  const float* wq = (const float*)d_in[1];  const float* bq = (const float*)d_in[2];
  const float* wk = (const float*)d_in[3];  const float* bk = (const float*)d_in[4];
  const float* wv = (const float*)d_in[5];  const float* bv = (const float*)d_in[6];
  const float* wo = (const float*)d_in[7];  const float* bo = (const float*)d_in[8];
  const float* w1 = (const float*)d_in[9];  const float* b1 = (const float*)d_in[10];
  const float* w2 = (const float*)d_in[11]; const float* b2 = (const float*)d_in[12];
  const float* ln1g = (const float*)d_in[13]; const float* ln1b = (const float*)d_in[14];
  const float* ln2g = (const float*)d_in[15]; const float* ln2b = (const float*)d_in[16];

  char* ws = (char*)d_ws;
  size_t off = 0;
  auto alloc = [&](size_t bytes) -> void* {
    void* p = ws + off;
    off += (bytes + 255) & ~(size_t)255;
    return p;
  };
  const size_t MD_F = (size_t)MTOK * DMODEL * sizeof(float);
  const size_t MD_B = (size_t)MTOK * DMODEL * sizeof(__bf16);
  const size_t W_B  = (size_t)KDIM * NDIM * sizeof(__bf16);

  float*  xf    = (float*) alloc(MD_F);
  __bf16* xb    = (__bf16*)alloc(MD_B);
  __bf16* wqt   = (__bf16*)alloc(W_B);
  __bf16* wkt   = (__bf16*)alloc(W_B);
  __bf16* wvt   = (__bf16*)alloc(W_B);
  __bf16* wot   = (__bf16*)alloc(W_B);
  __bf16* w1t   = (__bf16*)alloc(W_B);
  __bf16* w2t   = (__bf16*)alloc(W_B);
  __bf16* qh    = (__bf16*)alloc(MD_B);   // [B,H,S,64]
  __bf16* khd   = (__bf16*)alloc(MD_B);   // [B,H,S,64]
  __bf16* vth   = (__bf16*)alloc(MD_B);   // [B,H,64,S]
  __bf16* ctxb  = (__bf16*)alloc(MD_B);   // [B,S,D]
  float*  attnf = (float*) alloc(MD_F);
  float*  out1f = (float*) alloc(MD_F);
  __bf16* out1b = (__bf16*)alloc(MD_B);
  __bf16* hb    = (__bf16*)alloc(MD_B);
  float*  ffnf  = (float*) alloc(MD_F);

  dim3 blk(256);
  const int wBlocks = (KDIM * NDIM) / 256;
  transpose_w_bf16<<<wBlocks, blk, 0, stream>>>(wq, wqt);
  transpose_w_bf16<<<wBlocks, blk, 0, stream>>>(wk, wkt);
  transpose_w_bf16<<<wBlocks, blk, 0, stream>>>(wv, wvt);
  transpose_w_bf16<<<wBlocks, blk, 0, stream>>>(wo, wot);
  transpose_w_bf16<<<wBlocks, blk, 0, stream>>>(w1, w1t);
  transpose_w_bf16<<<wBlocks, blk, 0, stream>>>(w2, w2t);

  add_pe<<<(MTOK * DMODEL) / 256, blk, 0, stream>>>(inputs, xf, xb);

  dim3 gg(NDIM / 64, MTOK / 128);   // (16, 64)
  gemm_bf16_wmma<EPI_HEADS  ><<<gg, blk, 0, stream>>>(xb, wqt, bq, qh);
  gemm_bf16_wmma<EPI_HEADS  ><<<gg, blk, 0, stream>>>(xb, wkt, bk, khd);
  gemm_bf16_wmma<EPI_HEADS_T><<<gg, blk, 0, stream>>>(xb, wvt, bv, vth);

  attention_kernel<<<BATCH * NHEADS * (SEQ / 128), blk, 0, stream>>>(qh, khd, vth, ctxb);

  gemm_bf16_wmma<EPI_F32><<<gg, blk, 0, stream>>>(ctxb, wot, bo, attnf);
  residual_ln<<<MTOK, blk, 0, stream>>>(xf, attnf, ln1g, ln1b, out1f, out1b);

  gemm_bf16_wmma<EPI_RELU_BF16><<<gg, blk, 0, stream>>>(out1b, w1t, b1, hb);
  gemm_bf16_wmma<EPI_F32      ><<<gg, blk, 0, stream>>>(hb, w2t, b2, ffnf);

  residual_ln<<<MTOK, blk, 0, stream>>>(out1f, ffnf, ln2g, ln2b, (float*)d_out,
                                        (__bf16*)nullptr);
}
